// ProteinAutoEncoder_40836549050450
// MI455X (gfx1250) — matrix operand order repaired and checked by reading, not compiled, not run on hardware
//
#include <hip/hip_runtime.h>
#include <hip/hip_bf16.h>

typedef __attribute__((ext_vector_type(16))) _Float16 v16h;
typedef __attribute__((ext_vector_type(8)))  float    v8f;

#define BB      128
#define LL      2048
#define HD      8
#define NIN     37
#define NNODE   (BB * LL)        // 262144 chain nodes (reference chains across batch)
#define SEG     512              // valid nodes per workgroup
#define HALO    8                // 8 conv layers -> halo of 8
#define TOT     (SEG + 2 * HALO) // 528 local nodes
#define NTILE   (TOT / 16)       // 33 WMMA M-tiles
#define THREADS 256
#define NWAVES  (THREADS / 32)   // wave32: 8 waves per block

static_assert(TOT % 16 == 0, "tile alignment");

struct Params {
  const float *apos, *amask;
  const float *embW, *embB;
  const float *peW1, *peB1, *peW2, *peB2;
  const float *ceW1, *ceB1, *ceW2, *ceB2;
  const float *cpW1, *cpB1, *cpW2;
  const float *cnW1, *cnB1, *cnW2, *cnB2;
  const float *tW1, *tB1, *tW2, *tB2;
  const float *fW1, *fB1, *fW2, *fB2;
  const float *dW1, *dB1, *dW2, *dB2;
  const float *mW, *mB;
  float *out;
};

__device__ __forceinline__ float silu_f(float x) {
  // x * sigmoid(x) with fast v_rcp_f32 (no IEEE div sequence)
  return x * __builtin_amdgcn_rcpf(1.f + __expf(-x));
}

// Branchless B build for V_WMMA_F32_16X16X32_F16.
// B layout (16-bit 32x16): col = lane&15, K = 8*(lane>>4) + j (j<8), 16 + 8*(lane>>4) + (j-8) (j>=8).
__device__ __forceinline__ v16h make_b(const float* W, int KR, int NC) {
  const int lane = threadIdx.x & 31;
  const int col  = lane & 15;
  const int kb   = (lane >> 4) * 8;
  const int cc   = col < NC ? col : NC - 1;
  v16h b;
#pragma unroll
  for (int j = 0; j < 16; ++j) {
    const int k = j < 8 ? kb + j : 16 + kb + (j - 8);
    const float w = W[(k < KR ? k : KR - 1) * NC + cc];  // clamped, unconditional load
    b[j] = (_Float16)((k < KR && col < NC) ? w : 0.f);   // v_cndmask, no branch
  }
  return b;
}

// Per-lane bias (clamp+select, hoisted out of the tile loop).
__device__ __forceinline__ float ldbias(const float* b) {
  const int n = threadIdx.x & 15;
  const float v = b[n < HD ? n : HD - 1];
  return n < HD ? v : 0.f;
}

// Pack a contiguous 8-float LDS row into A elements 0..7 (K = 8*kh .. 8*kh+7),
// zeroing the kh=1 half via multiply (branchless; two ds_load_b128 per lane).
__device__ __forceinline__ v16h packA8(const float* row, int kh) {
  const float4* r = (const float4*)row;
  const float4 x0 = r[0];
  const float4 x1 = r[1];
  const float s = kh ? 0.f : 1.f;
  v16h a;
  a[0] = (_Float16)(x0.x * s); a[1] = (_Float16)(x0.y * s);
  a[2] = (_Float16)(x0.z * s); a[3] = (_Float16)(x0.w * s);
  a[4] = (_Float16)(x1.x * s); a[5] = (_Float16)(x1.y * s);
  a[6] = (_Float16)(x1.z * s); a[7] = (_Float16)(x1.w * s);
#pragma unroll
  for (int j = 8; j < 16; ++j) a[j] = (_Float16)0.f;
  return a;
}

// One wave computes 16(M) x 16(N) tiles. A layout (16-bit 16x32):
// lane = {row = lane&15, kh = lane>>4}; element j<8 -> K = j + 8*kh,
// element j>=8 -> K = 16 + (j-8) + 8*kh. D: lane holds col n = lane&15,
// rows r + 8*kh of the tile.
template <typename AV, typename EP>
__device__ __forceinline__ void gemm_core(v16h bm, AV abuild, EP ep) {
  const int lane = threadIdx.x & 31;
  const int wave = threadIdx.x >> 5;
  const int row  = lane & 15;
  const int kh   = lane >> 4;
  for (int t = wave; t < NTILE; t += NWAVES) {
    const v16h am = abuild(t * 16 + row, kh);
    v8f acc = {};
    acc = __builtin_amdgcn_wmma_f32_16x16x32_f16(false, am, false, bm,
                                                 (short)0, acc, false, false);
    ep(t * 16, row, kh, acc);
  }
}

__global__ __launch_bounds__(THREADS) void pae_fused(Params P) {
  __shared__ float sh[TOT * HD];    // node features h (rows 32B-contiguous)
  __shared__ float sp[TOT * 3];     // node positions
  __shared__ float sea[TOT * HD];   // edge activations ea (edge e: node e -> e+1)
  __shared__ float stmp[TOT * HD];  // scratch hidden
  __shared__ float sdp[TOT * 3];    // edge pos deltas dp

  const int tid = threadIdx.x;
  const long gbase = (long)blockIdx.x * SEG - HALO;

  auto edgeOK = [&](int e) -> bool {
    const long g = gbase + e;
    return (e >= 0) && (e < TOT - 1) && (g >= 0) && (g < (long)NNODE - 1);
  };

  // ------------------------- encode: atoms -> (h, pos) -------------------------
  for (int i = tid; i < TOT; i += THREADS) {
    const long g = gbase + i;
    float hsum[HD] = {0, 0, 0, 0, 0, 0, 0, 0};
    float ps0 = 0.f, ps1 = 0.f, ps2 = 0.f, ms = 0.f;
    const bool ok = (g >= 0 && g < (long)NNODE);
    if (ok) {
      const float* mrow = P.amask + g * NIN;
      const float* prow = P.apos + g * (NIN * 3);
      __builtin_prefetch(prow + NIN * 3, 0, 0);  // global_prefetch_b8 next record
      for (int a = 0; a < NIN; ++a) {
        const float m = mrow[a];
        ms += m;
        ps0 += m * prow[a * 3 + 0];
        ps1 += m * prow[a * 3 + 1];
        ps2 += m * prow[a * 3 + 2];
#pragma unroll
        for (int c = 0; c < HD; ++c) hsum[c] += m * P.embW[a * HD + c];
      }
    }
    const float inv = __builtin_amdgcn_rcpf(ms + 1e-8f);
    const float mp[3] = {ps0 * inv, ps1 * inv, ps2 * inv};
    float ph[HD];
#pragma unroll
    for (int j = 0; j < HD; ++j) {
      float v = P.peB1[j];
#pragma unroll
      for (int d = 0; d < 3; ++d) v += mp[d] * P.peW1[d * HD + j];
      ph[j] = silu_f(v);
    }
#pragma unroll
    for (int c = 0; c < HD; ++c) {
      float v = P.peB2[c] + P.embB[c] + hsum[c];
#pragma unroll
      for (int j = 0; j < HD; ++j) v += ph[j] * P.peW2[j * HD + c];
      sh[i * HD + c] = ok ? v : 0.f;
    }
    sp[i * 3 + 0] = mp[0];
    sp[i * 3 + 1] = mp[1];
    sp[i * 3 + 2] = mp[2];
  }
  __syncthreads();

  // pointwise H->H WMMA pass (tol / froml)
  auto mlp_pass = [&](const float* W, const float* bvec, bool act,
                      float* dst, const float* src) {
    const v16h bm = make_b(W, HD, HD);
    const float bias = ldbias(bvec);
    gemm_core(bm,
      [&](int e, int kh) -> v16h { return packA8(src + e * HD, kh); },
      [&](int mb, int n, int kh, v8f acc) {
        if (n < HD) {
          float* d = dst + (mb + kh * 8) * HD + n;
#pragma unroll
          for (int r = 0; r < 8; ++r)
            d[r * HD] = act ? silu_f(acc[r] + bias) : (acc[r] + bias);
        }
      });
    __syncthreads();
  };

  // ------------------------- 8 fused SE3 conv layers -------------------------
  for (int li = 0; li < 8; ++li) {
    if (li == 4) {  // bottleneck: tol then froml (pointwise, no halo cost)
      mlp_pass(P.tW1, P.tB1, true,  stmp, sh);
      mlp_pass(P.tW2, P.tB2, false, sh,   stmp);
      mlp_pass(P.fW1, P.fB1, true,  stmp, sh);
      mlp_pass(P.fW2, P.fB2, false, sh,   stmp);
    }
    const float* eW1 = P.ceW1 + li * (2 * HD + 1) * HD;
    const float* eB1 = P.ceB1 + li * HD;
    const float* eW2 = P.ceW2 + li * HD * HD;
    const float* eB2 = P.ceB2 + li * HD;
    const float* pW1 = P.cpW1 + li * HD * HD;
    const float* pB1 = P.cpB1 + li * HD;
    const float* pW2 = P.cpW2 + li * HD * 3;
    const float* nW1 = P.cnW1 + li * 2 * HD * HD;
    const float* nB1 = P.cnB1 + li * HD;
    const float* nW2 = P.cnW2 + li * HD * HD;
    const float* nB2 = P.cnB2 + li * HD;

    // pass 1: edge feature [h[e], h[e+1], dist] (K=17) -> silu hidden.
    // kh=0 lanes supply K 0..7 = h[e], kh=1 lanes K 8..15 = h[e+1]
    // -> same code path via row (e+kh); element 8 is dist for kh=0 only.
    {
      const v16h bm = make_b(eW1, 2 * HD + 1, HD);
      const float bias = ldbias(eB1);
      gemm_core(bm,
        [&](int e, int kh) -> v16h {
          if (e > TOT - 2) e = TOT - 2;  // keep e+1 in bounds (rim, discarded)
          const float4* r = (const float4*)(sh + (e + kh) * HD);
          const float4 x0 = r[0];
          const float4 x1 = r[1];
          const float dx = sp[(e + 1) * 3 + 0] - sp[e * 3 + 0];
          const float dy = sp[(e + 1) * 3 + 1] - sp[e * 3 + 1];
          const float dz = sp[(e + 1) * 3 + 2] - sp[e * 3 + 2];
          const float dist = sqrtf(dx * dx + dy * dy + dz * dz);
          const float tm = kh ? 0.f : 1.f;
          v16h a;
          a[0] = (_Float16)x0.x; a[1] = (_Float16)x0.y;
          a[2] = (_Float16)x0.z; a[3] = (_Float16)x0.w;
          a[4] = (_Float16)x1.x; a[5] = (_Float16)x1.y;
          a[6] = (_Float16)x1.z; a[7] = (_Float16)x1.w;
          a[8] = (_Float16)(dist * tm);
#pragma unroll
          for (int j = 9; j < 16; ++j) a[j] = (_Float16)0.f;
          return a;
        },
        [&](int mb, int n, int kh, v8f acc) {
          if (n < HD) {
            float* d = stmp + (mb + kh * 8) * HD + n;
#pragma unroll
            for (int r = 0; r < 8; ++r) d[r * HD] = silu_f(acc[r] + bias);
          }
        });
      __syncthreads();
    }

    // pass 2: hidden -> ea (no activation)
    mlp_pass(eW2, eB2, false, sea, stmp);

    // pass 3: ea -> silu pos-hidden
    mlp_pass(pW1, pB1, true, stmp, sea);

    // pass 4: pos-hidden -> dp (N=3, no bias)
    {
      const v16h bm = make_b(pW2, HD, 3);
      gemm_core(bm,
        [&](int e, int kh) -> v16h { return packA8(stmp + e * HD, kh); },
        [&](int mb, int n, int kh, v8f acc) {
          if (n < 3) {
            float* d = sdp + (mb + kh * 8) * 3 + n;
#pragma unroll
            for (int r = 0; r < 8; ++r) d[r * 3] = acc[r];
          }
        });
      __syncthreads();
    }

    // pos[i] += 0.1 * (dp[i] - dp[i-1])  (chain-end masking)
    for (int i = tid; i < TOT; i += THREADS) {
      const float m1 = edgeOK(i) ? 1.f : 0.f;
      const float m0 = edgeOK(i - 1) ? 1.f : 0.f;
      const int i0 = i > 0 ? i - 1 : 0;
#pragma unroll
      for (int d = 0; d < 3; ++d)
        sp[i * 3 + d] += 0.1f * (m1 * sdp[i * 3 + d] - m0 * sdp[i0 * 3 + d]);
    }
    __syncthreads();

    // pass 5: node mlp 1 — A = [h[i] | ea[i]+ea[i-1]] (K=16), node_upd on the fly
    {
      const v16h bm = make_b(nW1, 2 * HD, HD);
      const float bias = ldbias(nB1);
      gemm_core(bm,
        [&](int i, int kh) -> v16h {
          v16h a;
          if (kh == 0) {  // K 0..7 = h[i]
            const float4* r = (const float4*)(sh + i * HD);
            const float4 x0 = r[0];
            const float4 x1 = r[1];
            a[0] = (_Float16)x0.x; a[1] = (_Float16)x0.y;
            a[2] = (_Float16)x0.z; a[3] = (_Float16)x0.w;
            a[4] = (_Float16)x1.x; a[5] = (_Float16)x1.y;
            a[6] = (_Float16)x1.z; a[7] = (_Float16)x1.w;
          } else {        // K 8..15 = masked ea[i] + ea[i-1]
            const float m1 = edgeOK(i) ? 1.f : 0.f;
            const float m0 = edgeOK(i - 1) ? 1.f : 0.f;
            const int i1 = i < TOT - 1 ? i : TOT - 2;
            const int i0 = i > 0 ? i - 1 : 0;
            const float4* r1 = (const float4*)(sea + i1 * HD);
            const float4* r0 = (const float4*)(sea + i0 * HD);
            const float4 y0 = r1[0], y1 = r1[1];
            const float4 z0 = r0[0], z1 = r0[1];
            a[0] = (_Float16)(m1 * y0.x + m0 * z0.x);
            a[1] = (_Float16)(m1 * y0.y + m0 * z0.y);
            a[2] = (_Float16)(m1 * y0.z + m0 * z0.z);
            a[3] = (_Float16)(m1 * y0.w + m0 * z0.w);
            a[4] = (_Float16)(m1 * y1.x + m0 * z1.x);
            a[5] = (_Float16)(m1 * y1.y + m0 * z1.y);
            a[6] = (_Float16)(m1 * y1.z + m0 * z1.z);
            a[7] = (_Float16)(m1 * y1.w + m0 * z1.w);
          }
#pragma unroll
          for (int j = 8; j < 16; ++j) a[j] = (_Float16)0.f;
          return a;
        },
        [&](int mb, int n, int kh, v8f acc) {
          if (n < HD) {
            float* d = stmp + (mb + kh * 8) * HD + n;
#pragma unroll
            for (int r = 0; r < 8; ++r) d[r * HD] = silu_f(acc[r] + bias);
          }
        });
      __syncthreads();
    }

    // pass 6: node mlp 2 -> new h
    mlp_pass(nW2, nB2, false, sh, stmp);
  }

  // ------------------------- decode -------------------------
  // hidden = silu(h @ posdec_W1 + b1): 16 values -> stmp[0:8], sea[8:16]
  for (int i = tid; i < TOT; i += THREADS) {
    float hv[HD];
#pragma unroll
    for (int k = 0; k < HD; ++k) hv[k] = sh[i * HD + k];
#pragma unroll
    for (int j = 0; j < 2 * HD; ++j) {
      float v = P.dB1[j];
#pragma unroll
      for (int k = 0; k < HD; ++k) v += hv[k] * P.dW1[k * 2 * HD + j];
      v = silu_f(v);
      if (j < HD) stmp[i * HD + j] = v;
      else        sea[i * HD + (j - HD)] = v;
    }
  }
  __syncthreads();

  // coalesced output: consecutive threads write consecutive channels
  const int OC = NIN * 3 + NIN;  // 111 + 37 = 148 outputs per node
  const size_t PO = (size_t)NNODE * (NIN * 3);
  for (int idx = tid; idx < SEG * OC; idx += THREADS) {
    const int i = HALO + idx / OC;
    const int c = idx % OC;
    const long g = gbase + i;
    if (c < NIN * 3) {
      float v = P.dB2[c];
#pragma unroll
      for (int k = 0; k < HD; ++k) v += stmp[i * HD + k] * P.dW2[k * (NIN * 3) + c];
#pragma unroll
      for (int k = 0; k < HD; ++k) v += sea[i * HD + k] * P.dW2[(HD + k) * (NIN * 3) + c];
      P.out[(size_t)g * (NIN * 3) + c] = v;
    } else {
      const int c2 = c - NIN * 3;
      float v = P.mB[c2];
#pragma unroll
      for (int k = 0; k < HD; ++k) v += sh[i * HD + k] * P.mW[k * NIN + c2];
      P.out[PO + (size_t)g * NIN + c2] = v;
    }
  }
}

extern "C" void kernel_launch(void* const* d_in, const int* in_sizes, int n_in,
                              void* d_out, int out_size, void* d_ws, size_t ws_size,
                              hipStream_t stream) {
  (void)in_sizes; (void)n_in; (void)out_size; (void)d_ws; (void)ws_size;
  Params P;
  P.apos = (const float*)d_in[0];
  P.amask = (const float*)d_in[1];
  P.embW = (const float*)d_in[2];
  P.embB = (const float*)d_in[3];
  P.peW1 = (const float*)d_in[4];
  P.peB1 = (const float*)d_in[5];
  P.peW2 = (const float*)d_in[6];
  P.peB2 = (const float*)d_in[7];
  P.ceW1 = (const float*)d_in[8];
  P.ceB1 = (const float*)d_in[9];
  P.ceW2 = (const float*)d_in[10];
  P.ceB2 = (const float*)d_in[11];
  P.cpW1 = (const float*)d_in[12];
  P.cpB1 = (const float*)d_in[13];
  P.cpW2 = (const float*)d_in[14];
  P.cnW1 = (const float*)d_in[15];
  P.cnB1 = (const float*)d_in[16];
  P.cnW2 = (const float*)d_in[17];
  P.cnB2 = (const float*)d_in[18];
  P.tW1 = (const float*)d_in[19];
  P.tB1 = (const float*)d_in[20];
  P.tW2 = (const float*)d_in[21];
  P.tB2 = (const float*)d_in[22];
  P.fW1 = (const float*)d_in[23];
  P.fB1 = (const float*)d_in[24];
  P.fW2 = (const float*)d_in[25];
  P.fB2 = (const float*)d_in[26];
  P.dW1 = (const float*)d_in[27];
  P.dB1 = (const float*)d_in[28];
  P.dW2 = (const float*)d_in[29];
  P.dB2 = (const float*)d_in[30];
  P.mW = (const float*)d_in[31];
  P.mB = (const float*)d_in[32];
  P.out = (float*)d_out;
  pae_fused<<<NNODE / SEG, THREADS, 0, stream>>>(P);
}
